// InformerEncoderLayer_498216206475
// MI455X (gfx1250) — matrix operand order repaired and checked by reading, not compiled, not run on hardware
//
#include <hip/hip_runtime.h>
#include <math.h>

#define D_MODEL 1024
#define NHEADS  16
#define DK      64
#define D_FF    4096

typedef unsigned short u16;
typedef __attribute__((ext_vector_type(16))) __bf16 v16bf;
typedef __attribute__((ext_vector_type(8)))  float  v8f;

union FragBF {
    v16bf bf;
    uint4 u[2];
    u16   us[16];
};

__device__ __forceinline__ u16 f32_bf16(float f) {
    unsigned int u = __float_as_uint(f);
    unsigned int r = 0x7FFFu + ((u >> 16) & 1u);  // round to nearest even
    return (u16)((u + r) >> 16);
}

// Async copy 16 bytes/lane from global directly into LDS (gfx1250 path,
// tracked by ASYNCcnt). lds_off = byte offset into LDS.
__device__ __forceinline__ void async_copy_b128(unsigned lds_off, const void* g) {
    asm volatile("global_load_async_to_lds_b128 %0, %1, off"
                 :: "v"(lds_off), "v"((unsigned long long)g) : "memory");
}
__device__ __forceinline__ void wait_async0() {
    asm volatile("s_wait_asynccnt 0x0" ::: "memory");
}
// Previous tile's 4 async ops done; the 4 just-issued may still be in flight.
__device__ __forceinline__ void wait_async4() {
    asm volatile("s_wait_asynccnt 0x4" ::: "memory");
}
__device__ __forceinline__ unsigned lds_offset(const void* p) {
    return (unsigned)(unsigned long long)p;
}

// ---------------------------------------------------------------------------
// One-time conversions
// ---------------------------------------------------------------------------
__global__ __launch_bounds__(256) void convert_bf16_k(
    const float* __restrict__ in, u16* __restrict__ out)
{
    size_t i4 = ((size_t)blockIdx.x * 256 + threadIdx.x) * 4;
    float4 f = *(const float4*)(in + i4);
    ushort4 o;
    o.x = f32_bf16(f.x); o.y = f32_bf16(f.y);
    o.z = f32_bf16(f.z); o.w = f32_bf16(f.w);
    *(ushort4*)(out + i4) = o;
}

// Wt[n][k] = bf16(W[k][n]);  N = 1<<nShift
__global__ __launch_bounds__(256) void transpose_bf16_k(
    const float* __restrict__ W, u16* __restrict__ Wt, int K, int nShift)
{
    size_t idx = (size_t)blockIdx.x * 256 + threadIdx.x;
    int k = (int)(idx >> nShift);
    int n = (int)(idx & ((1u << nShift) - 1u));
    Wt[(size_t)n * K + k] = f32_bf16(W[idx]);
}

// ---------------------------------------------------------------------------
// GEMM: C[M,N] = act(A[M,K] @ W[K,N] + bias (+ res)). A bf16 [M][K],
// Wt bf16 [N][K] (pre-transposed). Double-buffered async-to-LDS pipeline:
// tile t+1 streams in while WMMAs consume tile t.
// Block 128x128, K-step 32, 256 threads = 8 waves (2x4 WMMA tiles/wave).
// ---------------------------------------------------------------------------
template <bool RELU, bool RES, bool OUTBF>
__global__ __launch_bounds__(256) void gemm_bf16_wmma(
    const u16* __restrict__ A, const u16* __restrict__ Wt,
    const float* __restrict__ bias, const float* __restrict__ res,
    void* __restrict__ Cout, int M, int N, int K)
{
    __shared__ u16 As[2][128 * 32];  // [m][k]
    __shared__ u16 Bs[2][128 * 32];  // [n][k]

    const int tid   = threadIdx.x;
    const int lane  = tid & 31;
    const int wid   = tid >> 5;
    const int waveM = wid & 3;
    const int waveN = wid >> 2;
    const int hl    = lane >> 4;
    const int l16   = lane & 15;

    const int bm = blockIdx.y * 128;
    const int bn = blockIdx.x * 128;

    const unsigned asBase = lds_offset(&As[0][0]);
    const unsigned bsBase = lds_offset(&Bs[0][0]);

    // Per-thread staging coords: 512 chunks of 16B per tile, 2 per thread.
    const int row0 = tid >> 2;                 // chunk = tid
    const int c0   = (tid & 3) * 8;
    const int row1 = (tid + 256) >> 2;         // chunk = tid + 256
    const int c1   = c0;                       // (chunk & 3) identical

    // Issue one tile's async copies (4 instructions per wave).
    auto issue_tile = [&](int buf, int k0) {
        unsigned bo = (unsigned)buf * (128 * 32 * 2);
        async_copy_b128(asBase + bo + (unsigned)(row0 * 32 + c0) * 2,
                        A + (size_t)(bm + row0) * K + k0 + c0);
        async_copy_b128(bsBase + bo + (unsigned)(row0 * 32 + c0) * 2,
                        Wt + (size_t)(bn + row0) * K + k0 + c0);
        async_copy_b128(asBase + bo + (unsigned)(row1 * 32 + c1) * 2,
                        A + (size_t)(bm + row1) * K + k0 + c1);
        async_copy_b128(bsBase + bo + (unsigned)(row1 * 32 + c1) * 2,
                        Wt + (size_t)(bn + row1) * K + k0 + c1);
    };

    v8f acc[2][4];
#pragma unroll
    for (int i = 0; i < 2; ++i)
#pragma unroll
        for (int j = 0; j < 4; ++j) acc[i][j] = v8f{};

    issue_tile(0, 0);

    for (int k0 = 0; k0 < K; k0 += 32) {
        const int buf = (k0 >> 5) & 1;
        if (k0 + 32 < K) {
            issue_tile(buf ^ 1, k0 + 32);
            wait_async4();                 // tile t complete; t+1 in flight
        } else {
            wait_async0();
        }
        __syncthreads();

        FragBF a[2], b[4];
#pragma unroll
        for (int i = 0; i < 2; ++i) {
            int row = waveM * 32 + i * 16 + l16;
            const u16* p = &As[buf][row * 32 + hl * 8];
            a[i].u[0] = *(const uint4*)(p);       // K = hl*8 .. +7
            a[i].u[1] = *(const uint4*)(p + 16);  // K = 16+hl*8 .. +7
        }
#pragma unroll
        for (int j = 0; j < 4; ++j) {
            int n = waveN * 64 + j * 16 + l16;
            const u16* p = &Bs[buf][n * 32 + hl * 8];
            b[j].u[0] = *(const uint4*)(p);       // K = hl*8 .. +7
            b[j].u[1] = *(const uint4*)(p + 16);  // K = 16+hl*8 .. +7
        }
#pragma unroll
        for (int i = 0; i < 2; ++i)
#pragma unroll
            for (int j = 0; j < 4; ++j)
                acc[i][j] = __builtin_amdgcn_wmma_f32_16x16x32_bf16(
                    false, a[i].bf, false, b[j].bf, (short)0, acc[i][j],
                    false, false);
        __syncthreads();  // all reads of buf done before it is refilled
    }

    // Epilogue: vgpr r -> M = r + 8*hl, N = l16
#pragma unroll
    for (int i = 0; i < 2; ++i) {
        int m0 = bm + waveM * 32 + i * 16 + 8 * hl;
#pragma unroll
        for (int j = 0; j < 4; ++j) {
            int n = bn + waveN * 64 + j * 16 + l16;
            float bv = bias[n];
#pragma unroll
            for (int r = 0; r < 8; ++r) {
                int m   = m0 + r;
                float v = acc[i][j][r] + bv;
                if (RES)  v += res[(size_t)m * N + n];
                if (RELU) v  = v > 0.f ? v : 0.f;
                if (OUTBF) ((u16*)Cout)[(size_t)m * N + n]   = f32_bf16(v);
                else       ((float*)Cout)[(size_t)m * N + n] = v;
            }
        }
    }
}

// ---------------------------------------------------------------------------
// Flash attention over bf16 Q/K/V, bf16 ctx out. One block per
// (b, h, 128-query tile), 4 waves, wave = 32 query rows, 32-key tiles.
// ---------------------------------------------------------------------------
__global__ __launch_bounds__(128) void flash_attn_wmma(
    const u16* __restrict__ Q, const u16* __restrict__ Kv,
    const u16* __restrict__ V, u16* __restrict__ O, int S)
{
    __shared__ u16 Ks[32 * 64];     // [t][d]  B-layout for QK^T
    __shared__ u16 Vs[64 * 32];     // [d][t]  B-layout for PV
    __shared__ u16 Ps[4][32 * 32];  // per-wave P tile [m][t]

    const int tid  = threadIdx.x;
    const int lane = tid & 31;
    const int w    = tid >> 5;
    const int hl   = lane >> 4;
    const int l16  = lane & 15;

    const int b  = blockIdx.z;
    const int h  = blockIdx.y;
    const int q0 = blockIdx.x * 128 + w * 32;

    const size_t baseRow = (size_t)b * S;
    const int    hOff    = h * DK;
    const float  scale   = 0.125f;  // 1/sqrt(64), applied to scores

    const unsigned ksBase = lds_offset(&Ks[0]);

    // Q fragments straight from global bf16
    FragBF qf[2][2];
#pragma unroll
    for (int i = 0; i < 2; ++i) {
        int row = q0 + i * 16 + l16;
        const u16* qp = Q + (baseRow + row) * D_MODEL + hOff;
#pragma unroll
        for (int p = 0; p < 2; ++p) {
            int d0 = p * 32 + hl * 8;
            qf[i][p].u[0] = *(const uint4*)(qp + d0);
            qf[i][p].u[1] = *(const uint4*)(qp + d0 + 16);
        }
    }

    float mrow[2][8], lrow[2][8];
    v8f oacc[2][4];
#pragma unroll
    for (int i = 0; i < 2; ++i) {
#pragma unroll
        for (int r = 0; r < 8; ++r) { mrow[i][r] = -1e30f; lrow[i][r] = 0.f; }
#pragma unroll
        for (int j = 0; j < 4; ++j) oacc[i][j] = v8f{};
    }

    for (int kt = 0; kt < S; kt += 32) {
        // K tile: async copy, rows are 128B contiguous
#pragma unroll
        for (int it = 0; it < 2; ++it) {
            int chunk = tid + it * 128;          // 256 chunks of 16B
            int t     = chunk >> 3;              // 8 chunks per row
            int c     = (chunk & 7) * 8;
            async_copy_b128(ksBase + (unsigned)(t * 64 + c) * 2,
                            Kv + (baseRow + kt + t) * D_MODEL + hOff + c);
        }
        // V tile transposed: bf16 shuffle through VGPRs (no conversion)
#pragma unroll
        for (int it = 0; it < 4; ++it) {
            int chunk = tid + it * 128;          // 512 chunks of 4 elems
            int t     = chunk >> 4;
            int c4    = (chunk & 15) * 4;
            ushort4 vv = *(const ushort4*)(V + (baseRow + kt + t) * D_MODEL + hOff + c4);
            Vs[(c4 + 0) * 32 + t] = vv.x;
            Vs[(c4 + 1) * 32 + t] = vv.y;
            Vs[(c4 + 2) * 32 + t] = vv.z;
            Vs[(c4 + 3) * 32 + t] = vv.w;
        }
        wait_async0();
        __syncthreads();

        // scores = Q @ K^T
        v8f sacc[2][2];
#pragma unroll
        for (int i = 0; i < 2; ++i)
#pragma unroll
            for (int j = 0; j < 2; ++j) sacc[i][j] = v8f{};

        FragBF kb[2][2];
#pragma unroll
        for (int j = 0; j < 2; ++j) {
            int n = j * 16 + l16;
#pragma unroll
            for (int p = 0; p < 2; ++p) {
                const u16* kp = &Ks[n * 64 + p * 32 + hl * 8];
                kb[j][p].u[0] = *(const uint4*)(kp);
                kb[j][p].u[1] = *(const uint4*)(kp + 16);
            }
        }
#pragma unroll
        for (int i = 0; i < 2; ++i)
#pragma unroll
            for (int j = 0; j < 2; ++j)
#pragma unroll
                for (int p = 0; p < 2; ++p)
                    sacc[i][j] = __builtin_amdgcn_wmma_f32_16x16x32_bf16(
                        false, qf[i][p].bf, false, kb[j][p].bf, (short)0,
                        sacc[i][j], false, false);

        // Online softmax per 16-lane row group
#pragma unroll
        for (int i = 0; i < 2; ++i) {
#pragma unroll
            for (int r = 0; r < 8; ++r) {
                float s0 = sacc[i][0][r] * scale;
                float s1 = sacc[i][1][r] * scale;
                float mx = fmaxf(s0, s1);
#pragma unroll
                for (int off = 8; off >= 1; off >>= 1)
                    mx = fmaxf(mx, __shfl_xor(mx, off, 32));
                float mnew  = fmaxf(mrow[i][r], mx);
                float alpha = __expf(mrow[i][r] - mnew);
                mrow[i][r]  = mnew;
                float p0 = __expf(s0 - mnew);
                float p1 = __expf(s1 - mnew);
                float ps = p0 + p1;
#pragma unroll
                for (int off = 8; off >= 1; off >>= 1)
                    ps += __shfl_xor(ps, off, 32);
                lrow[i][r] = lrow[i][r] * alpha + ps;
#pragma unroll
                for (int j = 0; j < 4; ++j) oacc[i][j][r] *= alpha;
                int mloc = i * 16 + r + 8 * hl;
                Ps[w][mloc * 32 + l16]      = f32_bf16(p0);
                Ps[w][mloc * 32 + 16 + l16] = f32_bf16(p1);
            }
        }
        __syncthreads();

        // ctx += P @ V
        FragBF pf[2], vf[4];
#pragma unroll
        for (int i = 0; i < 2; ++i) {
            const u16* pp = &Ps[w][(i * 16 + l16) * 32 + hl * 8];
            pf[i].u[0] = *(const uint4*)(pp);
            pf[i].u[1] = *(const uint4*)(pp + 16);
        }
#pragma unroll
        for (int j = 0; j < 4; ++j) {
            int n = j * 16 + l16;
            const u16* vp = &Vs[n * 32 + hl * 8];
            vf[j].u[0] = *(const uint4*)(vp);
            vf[j].u[1] = *(const uint4*)(vp + 16);
        }
#pragma unroll
        for (int i = 0; i < 2; ++i)
#pragma unroll
            for (int j = 0; j < 4; ++j)
                oacc[i][j] = __builtin_amdgcn_wmma_f32_16x16x32_bf16(
                    false, pf[i].bf, false, vf[j].bf, (short)0, oacc[i][j],
                    false, false);
        __syncthreads();
    }

#pragma unroll
    for (int i = 0; i < 2; ++i)
#pragma unroll
        for (int j = 0; j < 4; ++j)
#pragma unroll
            for (int r = 0; r < 8; ++r) {
                int m = q0 + i * 16 + r + 8 * hl;
                int d = j * 16 + l16;
                O[(baseRow + m) * D_MODEL + hOff + d] =
                    f32_bf16(oacc[i][j][r] / lrow[i][r]);
            }
}

// ---------------------------------------------------------------------------
// Row-wise LayerNorm over 1024 (f32 in, f32 out + optional bf16 copy).
// ---------------------------------------------------------------------------
__global__ __launch_bounds__(256) void layernorm_k(
    const float* __restrict__ X, const float* __restrict__ gamma,
    const float* __restrict__ beta, float* __restrict__ Y,
    u16* __restrict__ Yb)
{
    __shared__ float red[32];
    const int row = blockIdx.x;
    const float* x = X + (size_t)row * D_MODEL;
    float4 v = *(const float4*)(x + threadIdx.x * 4);
    float s  = v.x + v.y + v.z + v.w;
    float sq = v.x * v.x + v.y * v.y + v.z * v.z + v.w * v.w;
#pragma unroll
    for (int off = 16; off >= 1; off >>= 1) {
        s  += __shfl_xor(s, off, 32);
        sq += __shfl_xor(sq, off, 32);
    }
    int lane = threadIdx.x & 31, w = threadIdx.x >> 5;
    if (lane == 0) { red[w] = s; red[w + 8] = sq; }
    __syncthreads();
    if (threadIdx.x == 0) {
        float a = 0.f, q = 0.f;
#pragma unroll
        for (int i = 0; i < 8; ++i) { a += red[i]; q += red[i + 8]; }
        float mean = a * (1.f / 1024.f);
        float var  = q * (1.f / 1024.f) - mean * mean;
        red[16] = mean;
        red[17] = rsqrtf(var + 1e-5f);
    }
    __syncthreads();
    float mean = red[16], rs = red[17];
    float4 g = *(const float4*)(gamma + threadIdx.x * 4);
    float4 b = *(const float4*)(beta + threadIdx.x * 4);
    float4 o;
    o.x = (v.x - mean) * rs * g.x + b.x;
    o.y = (v.y - mean) * rs * g.y + b.y;
    o.z = (v.z - mean) * rs * g.z + b.z;
    o.w = (v.w - mean) * rs * g.w + b.w;
    *(float4*)(Y + (size_t)row * D_MODEL + threadIdx.x * 4) = o;
    if (Yb) {
        ushort4 ob;
        ob.x = f32_bf16(o.x); ob.y = f32_bf16(o.y);
        ob.z = f32_bf16(o.z); ob.w = f32_bf16(o.w);
        *(ushort4*)(Yb + (size_t)row * D_MODEL + threadIdx.x * 4) = ob;
    }
}

// ---------------------------------------------------------------------------
extern "C" void kernel_launch(void* const* d_in, const int* in_sizes, int n_in,
                              void* d_out, int out_size, void* d_ws,
                              size_t ws_size, hipStream_t stream)
{
    (void)in_sizes; (void)n_in; (void)out_size; (void)ws_size;
    const float* x   = (const float*)d_in[0];
    const float* Wq  = (const float*)d_in[1];
    const float* bq  = (const float*)d_in[2];
    const float* Wk  = (const float*)d_in[3];
    const float* bk  = (const float*)d_in[4];
    const float* Wv  = (const float*)d_in[5];
    const float* bv  = (const float*)d_in[6];
    const float* Wo  = (const float*)d_in[7];
    const float* bo  = (const float*)d_in[8];
    const float* W1  = (const float*)d_in[9];
    const float* b1  = (const float*)d_in[10];
    const float* W2  = (const float*)d_in[11];
    const float* b2  = (const float*)d_in[12];
    const float* g1  = (const float*)d_in[13];
    const float* be1 = (const float*)d_in[14];
    const float* g2  = (const float*)d_in[15];
    const float* be2 = (const float*)d_in[16];
    float* out = (float*)d_out;

    const int B = 4, S = 2048;
    const int M = B * S;                              // 8192
    const size_t A  = (size_t)M * D_MODEL;            // 8.39M
    const size_t D2 = (size_t)D_MODEL * D_MODEL;      // 1.05M
    const size_t DF = (size_t)D_MODEL * D_FF;         // 4.19M

    u16* xb  = (u16*)d_ws;
    u16* Wqt = xb  + A;
    u16* Wkt = Wqt + D2;
    u16* Wvt = Wkt + D2;
    u16* Wot = Wvt + D2;
    u16* W1t = Wot + D2;            // [4096][1024]
    u16* W2t = W1t + DF;            // [1024][4096]
    u16* Qb  = W2t + DF;
    u16* Kb  = Qb  + A;
    u16* Vb  = Kb  + A;
    u16* CTX = Vb  + A;
    u16* X1b = CTX + A;
    u16* FF1 = X1b + A;             // [M][4096] bf16
    float* Y1  = (float*)(FF1 + (size_t)M * D_FF);
    float* X1f = Y1 + A;
    float* Y2  = Y1;                // Y1 dead after LN1

    dim3 blk(256);

    // One-time conversions (weights transposed to [N][K] bf16)
    convert_bf16_k<<<dim3((unsigned)(A / 1024)), blk, 0, stream>>>(x, xb);
    transpose_bf16_k<<<dim3((unsigned)(D2 / 256)), blk, 0, stream>>>(Wq, Wqt, D_MODEL, 10);
    transpose_bf16_k<<<dim3((unsigned)(D2 / 256)), blk, 0, stream>>>(Wk, Wkt, D_MODEL, 10);
    transpose_bf16_k<<<dim3((unsigned)(D2 / 256)), blk, 0, stream>>>(Wv, Wvt, D_MODEL, 10);
    transpose_bf16_k<<<dim3((unsigned)(D2 / 256)), blk, 0, stream>>>(Wo, Wot, D_MODEL, 10);
    transpose_bf16_k<<<dim3((unsigned)(DF / 256)), blk, 0, stream>>>(W1, W1t, D_MODEL, 12);
    transpose_bf16_k<<<dim3((unsigned)(DF / 256)), blk, 0, stream>>>(W2, W2t, D_FF, 10);

    dim3 gD(D_MODEL / 128, M / 128);
    dim3 gF(D_FF / 128, M / 128);

    gemm_bf16_wmma<false, false, true><<<gD, blk, 0, stream>>>(xb, Wqt, bq, nullptr, Qb, M, D_MODEL, D_MODEL);
    gemm_bf16_wmma<false, false, true><<<gD, blk, 0, stream>>>(xb, Wkt, bk, nullptr, Kb, M, D_MODEL, D_MODEL);
    gemm_bf16_wmma<false, false, true><<<gD, blk, 0, stream>>>(xb, Wvt, bv, nullptr, Vb, M, D_MODEL, D_MODEL);

    dim3 ga(S / 128, NHEADS, B);
    flash_attn_wmma<<<ga, dim3(128), 0, stream>>>(Qb, Kb, Vb, CTX, S);

    gemm_bf16_wmma<false, true, false><<<gD, blk, 0, stream>>>(CTX, Wot, bo, x, Y1, M, D_MODEL, D_MODEL);
    layernorm_k<<<dim3(M), blk, 0, stream>>>(Y1, g1, be1, X1f, X1b);

    gemm_bf16_wmma<true, false, true><<<gF, blk, 0, stream>>>(X1b, W1t, b1, nullptr, FF1, M, D_FF, D_MODEL);
    gemm_bf16_wmma<false, true, false><<<gD, blk, 0, stream>>>(FF1, W2t, b2, X1f, Y2, M, D_MODEL, D_FF);
    layernorm_k<<<dim3(M), blk, 0, stream>>>(Y2, g2, be2, out, nullptr);
}
